// LinearAttention_80272938762841
// MI455X (gfx1250) — matrix operand order repaired and checked by reading, not compiled, not run on hardware
//
#include <hip/hip_runtime.h>

// ---------------------------------------------------------------------------
// LinearAttention on MI455X (gfx1250):
//   * bf16 WMMA (v_wmma_f32_16x16x32_bf16) for the dominant 103-GFLOP qkv GEMM
//   * double-buffered LDS pipeline; A tiles staged via GLOBAL_LOAD_ASYNC_TO_LDS
//     (ASYNCcnt) overlapping the WMMA clause
//   * algebraically-collapsed tail: the reference einsum pair is an outer
//     product of per-(b,h) row sums, so q/v are never materialized.
// ---------------------------------------------------------------------------

#define B_  4
#define L_  4096
#define C_  1024
#define H_  16
#define D_  64
#define M_TOT (B_ * L_)      // 16384
#define N_TOT (3 * C_)       // 3072
#define K_TOT C_             // 1024

typedef __attribute__((ext_vector_type(16))) __bf16 v16bf;
typedef __attribute__((ext_vector_type(8)))  float  v8f;

__device__ __forceinline__ unsigned short f32_to_bf16(float f) {
    unsigned u = __float_as_uint(f);
    u = (u + 0x7FFFu + ((u >> 16) & 1u)) >> 16;   // round-to-nearest-even
    return (unsigned short)u;
}
__device__ __forceinline__ float bf16_to_f32(unsigned short h) {
    return __uint_as_float(((unsigned)h) << 16);
}

// --- CDNA5 async global->LDS copy (16B per lane), tracked by ASYNCcnt -------
__device__ __forceinline__ void async_copy_b128(const unsigned short* gbase,
                                                unsigned goff_bytes,
                                                const unsigned short* lds_ptr) {
    // LDS byte address = low 32 bits of the generic pointer (ISA: LDS aperture
    // maps addr[31:0] to the LDS offset).
    unsigned lds = (unsigned)(unsigned long long)lds_ptr;
    asm volatile("global_load_async_to_lds_b128 %0, %1, %2"
                 :: "v"(lds), "v"(goff_bytes), "s"(gbase)
                 : "memory");
}
__device__ __forceinline__ void wait_async0() {
#if __has_builtin(__builtin_amdgcn_s_wait_asynccnt)
    __builtin_amdgcn_s_wait_asynccnt(0);
#else
    asm volatile("s_wait_asynccnt 0" ::: "memory");
#endif
}

// ------------------------------ K0: convert --------------------------------
__global__ __launch_bounds__(256) void convert_bf16(const float4* __restrict__ src,
                                                    ushort4* __restrict__ dst,
                                                    int n4) {
    for (int i = blockIdx.x * 256 + threadIdx.x; i < n4; i += gridDim.x * 256) {
        float4 f = src[i];
        ushort4 o;
        o.x = f32_to_bf16(f.x); o.y = f32_to_bf16(f.y);
        o.z = f32_to_bf16(f.z); o.w = f32_to_bf16(f.w);
        dst[i] = o;
    }
}

// ------------------------------ K1: qkv GEMM -------------------------------
// C[m,n] = sum_k xb[m,k] * wb[k,n]
// Fused epilogue:
//   n in [0,1024)      (q):  elu+1, 16-col reduce  -> atomicAdd qsum
//   n in [1024,2048)   (k):  elu+1                 -> bf16 store to kfeat
//   n in [2048,3072)   (v):  raw,   16-col reduce  -> atomicAdd vrow
#define BM 128
#define BN 128
#define BK 32
#define AS_STRIDE 40   // shorts; 80B rows keep 16B LDS accesses aligned
#define BS_STRIDE 40

__global__ __launch_bounds__(256) void qkv_gemm(const unsigned short* __restrict__ xb,
                                                const unsigned short* __restrict__ wb,
                                                unsigned short* __restrict__ kfeat,
                                                float* __restrict__ qsum,
                                                float* __restrict__ vrow) {
    __shared__ unsigned short As[2][BM * AS_STRIDE];
    __shared__ unsigned short Bs[2][BN * BS_STRIDE];   // transposed: Bs[n][k]

    const int tid  = threadIdx.x;
    const int wave = tid >> 5;
    const int lane = tid & 31;
    const int hl   = lane >> 4;     // lane half (0/1)
    const int ln   = lane & 15;
    const int wm   = wave & 3;      // 4 waves along M -> 32 rows each
    const int wn   = wave >> 2;     // 2 waves along N -> 64 cols each
    const int blockM = blockIdx.y * BM;
    const int blockN = blockIdx.x * BN;

    // A staging decomposition: 128 rows x 32 bf16 = 512 x 16B chunks
    const int arow = tid >> 2;            // 0..63 (+64 on second pass)
    const int ach  = (tid & 3) * 8;       // short offset within row: 0,8,16,24
    // B staging decomposition: 32 k-rows x 128 n-cols, 8 bf16 per thread-load
    const int bkk  = tid >> 4;            // 0..15 (+16 on second pass)
    const int bn0  = (tid & 15) * 8;

#define STAGE_A_ASYNC(K0, BUF)                                                   \
    {                                                                            \
        _Pragma("unroll")                                                        \
        for (int p = 0; p < 2; ++p) {                                            \
            int row = arow + p * 64;                                             \
            unsigned go = (unsigned)(((blockM + row) * K_TOT + (K0) + ach) * 2); \
            async_copy_b128(xb, go, &As[BUF][row * AS_STRIDE + ach]);            \
        }                                                                        \
    }

#define SCATTER_B(W, BUF, KROW)                                                  \
    {                                                                            \
        const unsigned short* e8 = (const unsigned short*)&(W);                  \
        _Pragma("unroll")                                                        \
        for (int e = 0; e < 8; ++e)                                              \
            Bs[BUF][(bn0 + e) * BS_STRIDE + (KROW)] = e8[e];                     \
    }

    v8f acc[2][4];
#pragma unroll
    for (int i = 0; i < 2; ++i)
#pragma unroll
        for (int j = 0; j < 4; ++j)
#pragma unroll
            for (int e = 0; e < 8; ++e) acc[i][j][e] = 0.f;

    union Frag { v16bf bf; uint4 q[2]; };

    // ---- prologue: stage tile 0 into buffer 0
    STAGE_A_ASYNC(0, 0)
    {
        uint4 w0 = *(const uint4*)&wb[(size_t)bkk * N_TOT + blockN + bn0];
        uint4 w1 = *(const uint4*)&wb[(size_t)(bkk + 16) * N_TOT + blockN + bn0];
        SCATTER_B(w0, 0, bkk)
        SCATTER_B(w1, 0, bkk + 16)
    }
    wait_async0();
    __syncthreads();

    const int NSTEP = K_TOT / BK;   // 32
    for (int step = 0; step < NSTEP; ++step) {
        const int cur = step & 1;
        const int nxt = cur ^ 1;
        const int k0  = step * BK;
        const bool more = (step + 1 < NSTEP);

        // ---- issue next tile's loads before the math (overlap with WMMA)
        uint4 w0, w1;
        if (more) {
            STAGE_A_ASYNC(k0 + BK, nxt)
            w0 = *(const uint4*)&wb[(size_t)(k0 + BK + bkk) * N_TOT + blockN + bn0];
            w1 = *(const uint4*)&wb[(size_t)(k0 + BK + bkk + 16) * N_TOT + blockN + bn0];
        }

        // ---- fragments (ISA 16-bit A 16x32 / B 32x16 layouts) + WMMA
        v16bf afrag[2], bfrag[4];
#pragma unroll
        for (int i = 0; i < 2; ++i) {
            int row = wm * 32 + i * 16 + ln;
            Frag ua;
            ua.q[0] = *(const uint4*)&As[cur][row * AS_STRIDE + 8 * hl];
            ua.q[1] = *(const uint4*)&As[cur][row * AS_STRIDE + 16 + 8 * hl];
            afrag[i] = ua.bf;
        }
#pragma unroll
        for (int j = 0; j < 4; ++j) {
            int n = wn * 64 + j * 16 + ln;
            Frag ub;
            ub.q[0] = *(const uint4*)&Bs[cur][n * BS_STRIDE + 8 * hl];
            ub.q[1] = *(const uint4*)&Bs[cur][n * BS_STRIDE + 16 + 8 * hl];
            bfrag[j] = ub.bf;
        }
#pragma unroll
        for (int i = 0; i < 2; ++i)
#pragma unroll
            for (int j = 0; j < 4; ++j)
                acc[i][j] = __builtin_amdgcn_wmma_f32_16x16x32_bf16(
                    false, afrag[i], false, bfrag[j], (short)0, acc[i][j], false, false);

        // ---- finish staging the next B tile into the other buffer
        if (more) {
            SCATTER_B(w0, nxt, bkk)
            SCATTER_B(w1, nxt, bkk + 16)
        }
        wait_async0();        // this wave's async A writes to As[nxt] done
        __syncthreads();      // everyone's writes visible before buffer swap
    }

    // ---- fused epilogue
#pragma unroll
    for (int i = 0; i < 2; ++i) {
#pragma unroll
        for (int j = 0; j < 4; ++j) {
            const int colBase = blockN + wn * 64 + j * 16;   // multiple of 16
            const int col     = colBase + ln;
            const int region  = col >> 10;                   // 0=q 1=k 2=v (tile-uniform)
            const int cIn     = col & (C_ - 1);
            const int hh      = cIn >> 6;                    // head (tile-uniform)
#pragma unroll
            for (int r = 0; r < 8; ++r) {
                const int m = blockM + wm * 32 + i * 16 + r + 8 * hl;
                float v = acc[i][j][r];
                float f = (region == 2) ? v : (v > 0.f ? v + 1.f : __expf(v));
                if (region == 1) {
                    kfeat[(size_t)m * C_ + cIn] = f32_to_bf16(f);
                } else {
                    float s = f;                              // reduce 16 head-cols
#pragma unroll
                    for (int off = 8; off > 0; off >>= 1) s += __shfl_xor(s, off, 16);
                    if (ln == 0) {
                        const int b = m >> 12, l = m & (L_ - 1);
                        float* dst = (region == 0) ? qsum : vrow;
                        atomicAdd(&dst[((b * H_ + hh) << 12) + l], s);
                    }
                }
            }
        }
    }
#undef STAGE_A_ASYNC
#undef SCATTER_B
}

// ------------------------------ K2: kvsum GEMV -----------------------------
// kvsum[b,h,d] = sum_n kfeat[b*L+n, h*64+d] * vrow[b,h,n]
__global__ __launch_bounds__(256) void kv_reduce(const unsigned short* __restrict__ kfeat,
                                                 const float* __restrict__ vrow,
                                                 float* __restrict__ kvsum) {
    const int bh = blockIdx.x;
    const int b = bh >> 4, h = bh & 15;
    const int t = threadIdx.x;
    const int d = t & 63, g = t >> 6;
    float acc = 0.f;
    for (int n = g; n < L_; n += 4)
        acc += bf16_to_f32(kfeat[(size_t)(b * L_ + n) * C_ + h * D_ + d]) *
               vrow[(bh << 12) + n];
    __shared__ float red[256];
    red[t] = acc;
    __syncthreads();
    if (g == 0)
        kvsum[bh * D_ + d] = red[d] + red[64 + d] + red[128 + d] + red[192 + d];
}

// ------------------------------ K3: Wk -------------------------------------
// Wk[b,h,c] = sum_d kvsum[b,h,d] * W_proj[h*64+d, c]
__global__ __launch_bounds__(256) void wk_kernel(const float* __restrict__ kvsum,
                                                 const float* __restrict__ Wproj,
                                                 float* __restrict__ Wk) {
    const int bh = blockIdx.x;
    const int h = bh & 15;
    const int c = blockIdx.y * 256 + threadIdx.x;
    __shared__ float kv[D_];
    if (threadIdx.x < D_) kv[threadIdx.x] = kvsum[bh * D_ + threadIdx.x];
    __syncthreads();
    float acc = 0.f;
#pragma unroll 8
    for (int d = 0; d < D_; ++d)
        acc += kv[d] * Wproj[(size_t)(h * D_ + d) * C_ + c];
    Wk[(size_t)bh * C_ + c] = acc;
}

// ------------------------------ K4: output ---------------------------------
// y[b*L+l, c] = b_proj[c] + sum_h qsum[b,h,l] * Wk[b,h,c]
__global__ __launch_bounds__(256) void out_kernel(const float* __restrict__ qsum,
                                                  const float* __restrict__ Wk,
                                                  const float* __restrict__ bproj,
                                                  float* __restrict__ y) {
    const int row = blockIdx.x;               // 0..16383
    const int b = row >> 12, l = row & (L_ - 1);
    __shared__ float qs[H_];
    if (threadIdx.x < H_) qs[threadIdx.x] = qsum[((b * H_ + threadIdx.x) << 12) + l];
    __syncthreads();
    for (int c = threadIdx.x; c < C_; c += 256) {
        float acc = bproj[c];
#pragma unroll
        for (int h = 0; h < H_; ++h)
            acc += qs[h] * Wk[(size_t)(b * H_ + h) * C_ + c];
        y[(size_t)row * C_ + c] = acc;
    }
}

// ------------------------------ launcher -----------------------------------
extern "C" void kernel_launch(void* const* d_in, const int* in_sizes, int n_in,
                              void* d_out, int out_size, void* d_ws, size_t ws_size,
                              hipStream_t stream) {
    (void)in_sizes; (void)n_in; (void)out_size; (void)ws_size;
    const float* x     = (const float*)d_in[0];
    const float* Wqkv  = (const float*)d_in[1];
    const float* Wproj = (const float*)d_in[2];
    const float* bproj = (const float*)d_in[3];
    float* y = (float*)d_out;

    char* ws = (char*)d_ws;
    // workspace layout (16B-aligned chunks), ~75.8 MB total
    unsigned short* xb    = (unsigned short*)(ws);                       // 32 MB
    unsigned short* wb    = (unsigned short*)(ws + (size_t)33554432);    //  6 MB
    unsigned short* kfeat = (unsigned short*)(ws + (size_t)39845888);    // 32 MB
    float* qsum  = (float*)(ws + (size_t)73400320);                      //  1 MB
    float* vrow  = (float*)(ws + (size_t)74448896);                      //  1 MB
    float* kvsum = (float*)(ws + (size_t)75497472);                      // 16 KB
    float* Wk    = (float*)(ws + (size_t)75513856);                      // 256 KB

    convert_bf16<<<2048, 256, 0, stream>>>((const float4*)x,    (ushort4*)xb,
                                           (M_TOT * K_TOT) / 4);
    convert_bf16<<<1024, 256, 0, stream>>>((const float4*)Wqkv, (ushort4*)wb,
                                           (K_TOT * N_TOT) / 4);
    hipMemsetAsync(qsum, 0, (size_t)2 * 1048576, stream);   // zeros qsum + vrow

    qkv_gemm<<<dim3(N_TOT / BN, M_TOT / BM), 256, 0, stream>>>(xb, wb, kfeat, qsum, vrow);
    kv_reduce<<<B_ * H_, 256, 0, stream>>>(kfeat, vrow, kvsum);
    wk_kernel<<<dim3(B_ * H_, C_ / 256), 256, 0, stream>>>(kvsum, Wproj, Wk);
    out_kernel<<<M_TOT, 256, 0, stream>>>(qsum, Wk, bproj, y);
}
